// DeepGCN_80101140071044
// MI455X (gfx1250) — compile-verified, matrix operand order checked by reading, not run on hardware
//
#include <hip/hip_runtime.h>

// ---------------------------------------------------------------------------
// DeepGCN forward for MI455X (gfx1250, wave32, WMMA).
//
// Matrix work mapped to v_wmma_f32_16x16x32_bf16 with hi/lo bf16 split
// (3 WMMAs per 32-wide k-step -> ~fp32 accuracy for KNN distance ordering).
// EdgeConv reformulated as dense GEMM:  h_k = A*xi + W2*xj_k,
//   A = W[:, :C] - W[:, C:],  W2 = W[:, C:]
// BN (training-mode batch stats) absorbs the conv bias; scale > 0 lets
// max_k / relu commute with BN, so we only need per-point max + channel stats.
// ---------------------------------------------------------------------------

#define BB    8
#define NN    2048
#define KNN_K 12
#define CIN_  6
#define CH    64
#define NBLK  14
#define EMB_  1024
#define FUSE_ (CH * NBLK)   // 896
#define BN_   (BB * NN)     // 16384

typedef __bf16 bf16_t;
typedef bf16_t v16bf __attribute__((ext_vector_type(16)));
typedef float  v8f   __attribute__((ext_vector_type(8)));

union BfVec { v16bf v; unsigned short u[16]; };

__device__ inline unsigned short f32_to_bf16_rne(float f) {
  unsigned int x = __float_as_uint(f);
  unsigned int r = x + 0x7FFFu + ((x >> 16) & 1u);
  return (unsigned short)(r >> 16);
}
__device__ inline float bf16_bits_to_f32(unsigned short h) {
  return __uint_as_float(((unsigned int)h) << 16);
}

// k-offset within a 32-wide step per ISA 16-bit A-matrix table (7.12.2)
__device__ inline int frag_k(int v, int laneHi) {
  return (v < 4 ? 2 * v : 16 + 2 * (v - 4)) + (laneHi ? 8 : 0);
}

// A (16x32) fragment from pre-split bf16 (u16) matrix; B layout is symmetric
// (lane -> N column, same K packing) so the same loader serves both operands.
__device__ inline void load_frag_u16(const unsigned short* __restrict__ src,
                                     int rowBase, int ld, int k0, int lane, v16bf& out) {
  BfVec t;
  int laneHi = (lane >> 4) & 1;
  const unsigned short* p = src + (size_t)(rowBase + (lane & 15)) * ld + k0;
#pragma unroll
  for (int v = 0; v < 8; ++v) {
    int k = frag_k(v, laneHi);
    t.u[2 * v]     = p[k];
    t.u[2 * v + 1] = p[k + 1];
  }
  out = t.v;
}

// Same but from f32 source, producing hi/lo split in-register.
__device__ inline void load_frag_f32_hilo(const float* __restrict__ src,
                                          int rowBase, int ld, int k0, int lane,
                                          v16bf& hi, v16bf& lo) {
  BfVec th, tl;
  int laneHi = (lane >> 4) & 1;
  const float* p = src + (size_t)(rowBase + (lane & 15)) * ld + k0;
#pragma unroll
  for (int v = 0; v < 8; ++v) {
    int k = frag_k(v, laneHi);
    float a0 = p[k], a1 = p[k + 1];
    unsigned short h0 = f32_to_bf16_rne(a0), h1 = f32_to_bf16_rne(a1);
    th.u[2 * v]     = h0;
    th.u[2 * v + 1] = h1;
    tl.u[2 * v]     = f32_to_bf16_rne(a0 - bf16_bits_to_f32(h0));
    tl.u[2 * v + 1] = f32_to_bf16_rne(a1 - bf16_bits_to_f32(h1));
  }
  hi = th.v; lo = tl.v;
}

// 3-term split-precision accumulate: A*B ~= Ah*Bh + Ah*Bl + Al*Bh
__device__ inline v8f wmma3(v16bf ah, v16bf al, v16bf bh, v16bf bl, v8f acc) {
  acc = __builtin_amdgcn_wmma_f32_16x16x32_bf16(false, ah, false, bh, (short)0, acc, false, false);
  acc = __builtin_amdgcn_wmma_f32_16x16x32_bf16(false, ah, false, bl, (short)0, acc, false, false);
  acc = __builtin_amdgcn_wmma_f32_16x16x32_bf16(false, al, false, bh, (short)0, acc, false, false);
  return acc;
}

// ---------------------------------------------------------------------------
// Elementwise / setup kernels
// ---------------------------------------------------------------------------

__global__ void zero_kernel(float* p, int n) {
  int i = blockIdx.x * blockDim.x + threadIdx.x;
  if (i < n) p[i] = 0.0f;
}

__global__ void transpose_bcn_bnc(const float* __restrict__ src, float* __restrict__ dst,
                                  int Bv, int Cv, int Nv) {
  int i = blockIdx.x * blockDim.x + threadIdx.x;
  if (i >= Bv * Cv * Nv) return;
  int n = i % Nv, c = (i / Nv) % Cv, b = i / (Nv * Cv);
  dst[(size_t)b * Nv * Cv + (size_t)n * Cv + c] = src[i];
}

__global__ void cvt_hilo_kernel(const float* __restrict__ src,
                                unsigned short* __restrict__ hi,
                                unsigned short* __restrict__ lo, int n) {
  int i = blockIdx.x * blockDim.x + threadIdx.x;
  if (i >= n) return;
  float f = src[i];
  unsigned short h = f32_to_bf16_rne(f);
  hi[i] = h;
  lo[i] = f32_to_bf16_rne(f - bf16_bits_to_f32(h));
}

__global__ void rowsq_kernel(const float* __restrict__ x, float* __restrict__ sq, int rows) {
  int r = blockIdx.x * blockDim.x + threadIdx.x;
  if (r >= rows) return;
  const float* p = x + (size_t)r * CH;
  float s = 0.f;
#pragma unroll
  for (int c = 0; c < CH; ++c) s += p[c] * p[c];
  sq[r] = s;
}

// Wcomb (128x64): rows 0..63 = A = W1 - W2 ; rows 64..127 = W2
__global__ void make_wcomb_kernel(const float* __restrict__ Wb, float* __restrict__ Wc) {
  int i = blockIdx.x * blockDim.x + threadIdx.x;
  if (i >= CH * CH) return;
  int o = i >> 6, c = i & 63;
  float w1 = Wb[o * 2 * CH + c];
  float w2 = Wb[o * 2 * CH + CH + c];
  Wc[o * CH + c]        = w1 - w2;
  Wc[(CH + o) * CH + c] = w2;
}

// Head u/v: in-dim 6 is too small for WMMA; scalar is free here.
__global__ void head_uv_kernel(const float* __restrict__ xf, const float* __restrict__ Wh,
                               float* __restrict__ uv) {
  int i = blockIdx.x * blockDim.x + threadIdx.x;
  if (i >= BN_ * CH) return;
  int o = i & 63, bn = i >> 6;
  const float* x = xf + (size_t)bn * CIN_;
  float u = 0.f, v = 0.f;
#pragma unroll
  for (int c = 0; c < CIN_; ++c) {
    float w1 = Wh[o * 2 * CIN_ + c], w2 = Wh[o * 2 * CIN_ + CIN_ + c];
    u += (w1 - w2) * x[c];
    v += w2 * x[c];
  }
  uv[(size_t)bn * 128 + o]      = u;
  uv[(size_t)bn * 128 + 64 + o] = v;
}

// ---------------------------------------------------------------------------
// WMMA GEMM kernels (one 16x16 tile per wave, 8 waves per block)
// ---------------------------------------------------------------------------

__global__ void gemm_u16_kernel(const unsigned short* __restrict__ Ahi,
                                const unsigned short* __restrict__ Alo,
                                const unsigned short* __restrict__ Bhi,
                                const unsigned short* __restrict__ Blo,
                                float* __restrict__ C,
                                int M, int Nout, int Kdim, int lda, int ldb, int ldc) {
  int lane = threadIdx.x & 31;
  int tile = blockIdx.x * (blockDim.x >> 5) + (threadIdx.x >> 5);
  int tilesJ = Nout >> 4;
  if (tile >= (M >> 4) * tilesJ) return;            // wave-uniform guard
  int i0 = (tile / tilesJ) << 4;
  int j0 = (tile % tilesJ) << 4;
  v8f acc = {};
  for (int k0 = 0; k0 < Kdim; k0 += 32) {
    v16bf ah, al, bh, bl;
    load_frag_u16(Ahi, i0, lda, k0, lane, ah);
    load_frag_u16(Alo, i0, lda, k0, lane, al);
    load_frag_u16(Bhi, j0, ldb, k0, lane, bh);
    load_frag_u16(Blo, j0, ldb, k0, lane, bl);
    acc = wmma3(ah, al, bh, bl, acc);
  }
  int col = j0 + (lane & 15);
  int rbase = i0 + (((lane >> 4) & 1) << 3);
#pragma unroll
  for (int r = 0; r < 8; ++r) C[(size_t)(rbase + r) * ldc + col] = acc[r];
}

__global__ void gemm_f32a_u16b_kernel(const float* __restrict__ A,
                                      const unsigned short* __restrict__ Bhi,
                                      const unsigned short* __restrict__ Blo,
                                      float* __restrict__ C,
                                      int M, int Nout, int Kdim, int lda, int ldb, int ldc) {
  int lane = threadIdx.x & 31;
  int tile = blockIdx.x * (blockDim.x >> 5) + (threadIdx.x >> 5);
  int tilesJ = Nout >> 4;
  if (tile >= (M >> 4) * tilesJ) return;
  int i0 = (tile / tilesJ) << 4;
  int j0 = (tile % tilesJ) << 4;
  v8f acc = {};
  for (int k0 = 0; k0 < Kdim; k0 += 32) {
    v16bf ah, al, bh, bl;
    load_frag_f32_hilo(A, i0, lda, k0, lane, ah, al);
    load_frag_u16(Bhi, j0, ldb, k0, lane, bh);
    load_frag_u16(Blo, j0, ldb, k0, lane, bl);
    acc = wmma3(ah, al, bh, bl, acc);
  }
  int col = j0 + (lane & 15);
  int rbase = i0 + (((lane >> 4) & 1) << 3);
#pragma unroll
  for (int r = 0; r < 8; ++r) C[(size_t)(rbase + r) * ldc + col] = acc[r];
}

// dist[i][j] = sq[i] + sq[j] - 2 * <x_i, x_j>  (one batch; gram via WMMA)
__global__ void gram_dist_kernel(const unsigned short* __restrict__ Xhi,
                                 const unsigned short* __restrict__ Xlo,
                                 const float* __restrict__ sq,
                                 float* __restrict__ dist, int n) {
  int lane = threadIdx.x & 31;
  int tile = blockIdx.x * (blockDim.x >> 5) + (threadIdx.x >> 5);
  int tilesJ = n >> 4;
  if (tile >= tilesJ * tilesJ) return;
  int i0 = (tile / tilesJ) << 4;
  int j0 = (tile % tilesJ) << 4;
  v8f acc = {};
  for (int k0 = 0; k0 < CH; k0 += 32) {
    v16bf ah, al, bh, bl;
    load_frag_u16(Xhi, i0, CH, k0, lane, ah);
    load_frag_u16(Xlo, i0, CH, k0, lane, al);
    load_frag_u16(Xhi, j0, CH, k0, lane, bh);  // B = X^T : row base = j
    load_frag_u16(Xlo, j0, CH, k0, lane, bl);
    acc = wmma3(ah, al, bh, bl, acc);
  }
  int col = j0 + (lane & 15);
  int rbase = i0 + (((lane >> 4) & 1) << 3);
  float sqj = sq[col];
#pragma unroll
  for (int r = 0; r < 8; ++r) {
    int row = rbase + r;
    dist[(size_t)row * n + col] = sq[row] + sqj - 2.0f * acc[r];
  }
}

// ---------------------------------------------------------------------------
// KNN selection: per-row max-heap of k*d smallest (dist, idx), lexicographic
// tie-break on smaller index (matches jax.lax.top_k), then every d-th.
// ---------------------------------------------------------------------------

__device__ inline bool lex_gt(float d1, int i1, float d2, int i2) {
  return (d1 > d2) || (d1 == d2 && i1 > i2);
}
__device__ inline void sift_down(float* hd, int* hix, int start, int end) {
  int root = start;
  for (;;) {
    int ch = 2 * root + 1;
    if (ch >= end) break;
    if (ch + 1 < end && lex_gt(hd[ch + 1], hix[ch + 1], hd[ch], hix[ch])) ++ch;
    if (lex_gt(hd[ch], hix[ch], hd[root], hix[root])) {
      float td = hd[root]; hd[root] = hd[ch]; hd[ch] = td;
      int   ti = hix[root]; hix[root] = hix[ch]; hix[ch] = ti;
      root = ch;
    } else break;
  }
}

__global__ void knn_select_kernel(const float* __restrict__ dist, int* __restrict__ idxOut,
                                  int n, int k, int d) {
  int row = blockIdx.x * blockDim.x + threadIdx.x;
  if (row >= n) return;
  const int kd = k * d;                 // <= 156
  float hd[160]; int hix[160];
  const float* dr = dist + (size_t)row * n;
  for (int j = 0; j < kd; ++j) { hd[j] = dr[j]; hix[j] = j; }
  for (int s = kd / 2 - 1; s >= 0; --s) sift_down(hd, hix, s, kd);
  for (int j = kd; j < n; ++j) {
    float v = dr[j];
    if (!lex_gt(v, j, hd[0], hix[0])) { hd[0] = v; hix[0] = j; sift_down(hd, hix, 0, kd); }
  }
  for (int e = kd - 1; e > 0; --e) {    // heapsort -> ascending
    float td = hd[0]; hd[0] = hd[e]; hd[e] = td;
    int   ti = hix[0]; hix[0] = hix[e]; hix[e] = ti;
    sift_down(hd, hix, 0, e);
  }
  for (int t = 0; t < k; ++t) idxOut[(size_t)row * k + t] = hix[t * d];
}

// Head KNN on 3-D coords, d=1: stream distances, heap of 12.
__global__ void knn_head_kernel(const float* __restrict__ pts, int* __restrict__ idxOut, int n) {
  int b = blockIdx.y;
  int row = blockIdx.x * blockDim.x + threadIdx.x;
  if (row >= n) return;
  const float* pb = pts + (size_t)b * n * 3;
  float x0 = pb[row * 3], x1 = pb[row * 3 + 1], x2 = pb[row * 3 + 2];
  float sqi = x0 * x0 + x1 * x1 + x2 * x2;
  float hd[KNN_K]; int hix[KNN_K];
  for (int j = 0; j < KNN_K; ++j) {
    float y0 = pb[j * 3], y1 = pb[j * 3 + 1], y2 = pb[j * 3 + 2];
    hd[j] = sqi + y0 * y0 + y1 * y1 + y2 * y2 - 2.f * (x0 * y0 + x1 * y1 + x2 * y2);
    hix[j] = j;
  }
  for (int s = KNN_K / 2 - 1; s >= 0; --s) sift_down(hd, hix, s, KNN_K);
  for (int j = KNN_K; j < n; ++j) {
    float y0 = pb[j * 3], y1 = pb[j * 3 + 1], y2 = pb[j * 3 + 2];
    float v = sqi + y0 * y0 + y1 * y1 + y2 * y2 - 2.f * (x0 * y0 + x1 * y1 + x2 * y2);
    if (!lex_gt(v, j, hd[0], hix[0])) { hd[0] = v; hix[0] = j; sift_down(hd, hix, 0, KNN_K); }
  }
  for (int e = KNN_K - 1; e > 0; --e) {
    float td = hd[0]; hd[0] = hd[e]; hd[e] = td;
    int   ti = hix[0]; hix[0] = hix[e]; hix[e] = ti;
    sift_down(hd, hix, 0, e);
  }
  int* out = idxOut + (size_t)b * n * KNN_K + (size_t)row * KNN_K;
  for (int t = 0; t < KNN_K; ++t) out[t] = hix[t];
}

// ---------------------------------------------------------------------------
// EdgeConv epilogue: m = max_k (u + v[idx_k]); accumulate channel sum/sumsq.
// Block = 256 threads = 4 points x 64 channels.
// ---------------------------------------------------------------------------

__global__ void gather_max_stats_kernel(const float* __restrict__ uv,
                                        const int* __restrict__ idx,
                                        float* __restrict__ m, float* __restrict__ stats,
                                        int nPerB) {
  __shared__ float ls[CH], lss[CH];
  int t = threadIdx.x;
  if (t < CH) { ls[t] = 0.f; lss[t] = 0.f; }
  __syncthreads();
  int o = t & 63;
  int bn = blockIdx.x * 4 + (t >> 6);
  float s = 0.f, ss = 0.f;
  if (bn < BN_) {
    int b = bn / nPerB;
    float u = uv[(size_t)bn * 128 + o];
    float best = -3.0e38f;
    const int* ip = idx + (size_t)bn * KNN_K;
#pragma unroll
    for (int kk = 0; kk < KNN_K; ++kk) {
      int j = ip[kk];
      float h = u + uv[((size_t)(b * nPerB + j)) * 128 + 64 + o];
      best = fmaxf(best, h);
      s += h; ss += h * h;
    }
    m[(size_t)bn * CH + o] = best;
  }
  atomicAdd(&ls[o], s);
  atomicAdd(&lss[o], ss);
  __syncthreads();
  if (t < CH) { atomicAdd(&stats[t], ls[t]); atomicAdd(&stats[CH + t], lss[t]); }
}

// relu(BN(m)) (+ residual); writes next-block input and the f-concat slice.
__global__ void bn_apply_kernel(const float* __restrict__ m, const float* __restrict__ stats,
                                const float* __restrict__ g, const float* __restrict__ beta,
                                const float* __restrict__ resid,
                                float* __restrict__ Hout, float* __restrict__ fslice,
                                float cntInv) {
  int i = blockIdx.x * blockDim.x + threadIdx.x;
  if (i >= BN_ * CH) return;
  int o = i & 63, bn = i >> 6;
  float mean = stats[o] * cntInv;
  float var  = stats[CH + o] * cntInv - mean * mean;
  float sc = g[o] * rsqrtf(var + 1e-5f);
  float val = (m[i] - mean) * sc + beta[o];
  val = fmaxf(val, 0.f);
  if (resid) val += resid[i];
  Hout[i] = val;
  fslice[(size_t)bn * FUSE_ + o] = val;
}

// ---------------------------------------------------------------------------
// Fusion stats / pooling and MLP head
// ---------------------------------------------------------------------------

__global__ void colstats_kernel(const float* __restrict__ y, float* __restrict__ stats,
                                int rows, int cols) {
  __shared__ float bs[256], bss[256];
  int o = blockIdx.x;
  float s = 0.f, ss = 0.f;
  for (int r = threadIdx.x; r < rows; r += blockDim.x) {
    float v = y[(size_t)r * cols + o];
    s += v; ss += v * v;
  }
  bs[threadIdx.x] = s; bss[threadIdx.x] = ss;
  __syncthreads();
  for (int st = 128; st > 0; st >>= 1) {
    if ((int)threadIdx.x < st) { bs[threadIdx.x] += bs[threadIdx.x + st]; bss[threadIdx.x] += bss[threadIdx.x + st]; }
    __syncthreads();
  }
  if (threadIdx.x == 0) { stats[o] = bs[0]; stats[cols + o] = bss[0]; }
}

__global__ void fusion_reduce_kernel(const float* __restrict__ y, const float* __restrict__ stats,
                                     const float* __restrict__ g, const float* __restrict__ beta,
                                     float* __restrict__ z, int nPerB, int cols) {
  int t = blockIdx.x * blockDim.x + threadIdx.x;
  if (t >= BB * cols) return;
  int b = t / cols, o = t % cols;
  float cntInv = 1.0f / (float)(BB * nPerB);
  float mean = stats[o] * cntInv;
  float var  = stats[cols + o] * cntInv - mean * mean;
  float sc = g[o] * rsqrtf(var + 1e-5f), bet = beta[o];
  float mx = -3.0e38f, sum = 0.f;
  const float* p = y + (size_t)b * nPerB * cols + o;
  for (int n = 0; n < nPerB; ++n) {
    float v = (p[(size_t)n * cols] - mean) * sc + bet;
    v = v > 0.f ? v : 0.2f * v;
    mx = fmaxf(mx, v); sum += v;
  }
  z[(size_t)b * 2 * cols + o]        = mx;
  z[(size_t)b * 2 * cols + cols + o] = sum / (float)nPerB;
}

__global__ void fc_kernel(const float* __restrict__ x, const float* __restrict__ W,
                          const float* __restrict__ bias, float* __restrict__ out,
                          int rows, int inDim, int outDim) {
  int t = blockIdx.x * blockDim.x + threadIdx.x;
  if (t >= rows * outDim) return;
  int r = t / outDim, o = t % outDim;
  const float* xr = x + (size_t)r * inDim;
  const float* wr = W + (size_t)o * inDim;
  float acc = bias ? bias[o] : 0.f;
  for (int k = 0; k < inDim; ++k) acc += xr[k] * wr[k];
  out[t] = acc;
}

__global__ void bn_rows_leaky_kernel(float* __restrict__ x, const float* __restrict__ g,
                                     const float* __restrict__ beta, int rows, int cols) {
  int o = blockIdx.x * blockDim.x + threadIdx.x;
  if (o >= cols) return;
  float s = 0.f, ss = 0.f;
  for (int r = 0; r < rows; ++r) { float v = x[(size_t)r * cols + o]; s += v; ss += v * v; }
  float mean = s / rows, var = ss / rows - mean * mean;
  float sc = g[o] * rsqrtf(var + 1e-5f);
  for (int r = 0; r < rows; ++r) {
    float v = (x[(size_t)r * cols + o] - mean) * sc + beta[o];
    x[(size_t)r * cols + o] = v > 0.f ? v : 0.2f * v;
  }
}

// ---------------------------------------------------------------------------
// Host orchestration
// ---------------------------------------------------------------------------

extern "C" void kernel_launch(void* const* d_in, const int* in_sizes, int n_in,
                              void* d_out, int out_size, void* d_ws, size_t ws_size,
                              hipStream_t stream) {
  (void)in_sizes; (void)n_in; (void)out_size; (void)ws_size;

  const float* points   = (const float*)d_in[0];
  const float* features = (const float*)d_in[1];
  const float* W_head   = (const float*)d_in[2];
  const float* g_head   = (const float*)d_in[3];
  const float* beta_head= (const float*)d_in[4];
  const float* Wb       = (const float*)d_in[5];
  const float* gb       = (const float*)d_in[7];
  const float* betab    = (const float*)d_in[8];
  const float* Wf       = (const float*)d_in[9];
  const float* gf       = (const float*)d_in[10];
  const float* betaf    = (const float*)d_in[11];
  const float* W1       = (const float*)d_in[12];
  const float* b1       = (const float*)d_in[13];
  const float* g1       = (const float*)d_in[14];
  const float* beta1    = (const float*)d_in[15];
  const float* W2       = (const float*)d_in[16];
  const float* b2       = (const float*)d_in[17];
  const float* g2       = (const float*)d_in[18];
  const float* beta2    = (const float*)d_in[19];
  const float* W3       = (const float*)d_in[20];
  const float* b3       = (const float*)d_in[21];
  const float* Wp       = (const float*)d_in[22];
  const float* bp       = (const float*)d_in[23];
  float* out_f = (float*)d_out;

  char* wsp = (char*)d_ws;
  size_t off = 0;
  auto alloc = [&](size_t bytes) -> void* {
    void* p = wsp + off;
    off = (off + bytes + 255) & ~(size_t)255;
    return p;
  };

  float* f_concat = (float*)alloc((size_t)BN_ * FUSE_ * 4);     // 58.7 MB
  float* Hcur     = (float*)alloc((size_t)BN_ * CH * 4);
  float* m_buf    = (float*)alloc((size_t)BN_ * CH * 4);
  float* uv       = (float*)alloc((size_t)BN_ * 128 * 4);
  float* distb    = (float*)alloc((size_t)NN * NN * 4);         // per-batch reuse
  float* ybuf     = (float*)alloc((size_t)BN_ * EMB_ * 4);      // 64 MB
  float* pts_t    = (float*)alloc((size_t)BN_ * 3 * 4);
  float* xf       = (float*)alloc((size_t)BN_ * CIN_ * 4);
  float* sqb      = (float*)alloc((size_t)BN_ * 4);
  unsigned short* Hhi = (unsigned short*)alloc((size_t)BN_ * CH * 2);
  unsigned short* Hlo = (unsigned short*)alloc((size_t)BN_ * CH * 2);
  float* Wc       = (float*)alloc((size_t)128 * CH * 4);
  unsigned short* Wchi = (unsigned short*)alloc((size_t)128 * CH * 2);
  unsigned short* Wclo = (unsigned short*)alloc((size_t)128 * CH * 2);
  unsigned short* Wfhi = (unsigned short*)alloc((size_t)EMB_ * FUSE_ * 2);
  unsigned short* Wflo = (unsigned short*)alloc((size_t)EMB_ * FUSE_ * 2);
  int*   idxb     = (int*)alloc((size_t)BN_ * KNN_K * 4);
  float* stats64  = (float*)alloc(2 * CH * 4);
  float* statsF   = (float*)alloc(2 * EMB_ * 4);
  float* zvec     = (float*)alloc((size_t)BB * 2 * EMB_ * 4);
  float* t1       = (float*)alloc((size_t)BB * 512 * 4);
  float* t2       = (float*)alloc((size_t)BB * 256 * 4);

  const float cntInvEdge = 1.0f / (float)(BN_ * KNN_K);

  // --- input transposes ---
  transpose_bcn_bnc<<<(BB * 3 * NN + 255) / 256, 256, 0, stream>>>(points, pts_t, BB, 3, NN);
  transpose_bcn_bnc<<<(BB * CIN_ * NN + 255) / 256, 256, 0, stream>>>(features, xf, BB, CIN_, NN);

  // --- head block: KNN on coords, EdgeConv(6->64), no bias, no residual ---
  knn_head_kernel<<<dim3(NN / 64, BB), 64, 0, stream>>>(pts_t, idxb, NN);
  head_uv_kernel<<<(BN_ * CH) / 256, 256, 0, stream>>>(xf, W_head, uv);
  zero_kernel<<<1, 128, 0, stream>>>(stats64, 2 * CH);
  gather_max_stats_kernel<<<BN_ / 4, 256, 0, stream>>>(uv, idxb, m_buf, stats64, NN);
  bn_apply_kernel<<<(BN_ * CH) / 256, 256, 0, stream>>>(m_buf, stats64, g_head, beta_head,
                                                        nullptr, Hcur, f_concat, cntInvEdge);

  // --- 13 residual dynamic blocks ---
  for (int blk = 0; blk < NBLK - 1; ++blk) {
    // bf16 hi/lo split of current features + row norms (for gram -> dist)
    cvt_hilo_kernel<<<(BN_ * CH) / 256, 256, 0, stream>>>(Hcur, Hhi, Hlo, BN_ * CH);
    rowsq_kernel<<<BN_ / 256, 256, 0, stream>>>(Hcur, sqb, BN_);

    // per-batch dense KNN: WMMA gram -> dist -> heap top-(k*d) -> dilate
    for (int b = 0; b < BB; ++b) {
      gram_dist_kernel<<<(NN / 16) * (NN / 16) / 8, 256, 0, stream>>>(
          Hhi + (size_t)b * NN * CH, Hlo + (size_t)b * NN * CH, sqb + (size_t)b * NN, distb, NN);
      knn_select_kernel<<<NN / 64, 64, 0, stream>>>(distb, idxb + (size_t)b * NN * KNN_K,
                                                    NN, KNN_K, blk + 1);
    }

    // EdgeConv as one GEMM: [u|v] = H @ [A;W2]^T  (16384x64 @ 64x128)
    make_wcomb_kernel<<<(CH * CH) / 256, 256, 0, stream>>>(Wb + (size_t)blk * CH * 2 * CH, Wc);
    cvt_hilo_kernel<<<(128 * CH) / 256, 256, 0, stream>>>(Wc, Wchi, Wclo, 128 * CH);
    gemm_u16_kernel<<<((BN_ / 16) * (128 / 16)) / 8, 256, 0, stream>>>(
        Hhi, Hlo, Wchi, Wclo, uv, BN_, 128, CH, CH, CH, 128);

    zero_kernel<<<1, 128, 0, stream>>>(stats64, 2 * CH);
    gather_max_stats_kernel<<<BN_ / 4, 256, 0, stream>>>(uv, idxb, m_buf, stats64, NN);
    bn_apply_kernel<<<(BN_ * CH) / 256, 256, 0, stream>>>(
        m_buf, stats64, gb + (size_t)blk * CH, betab + (size_t)blk * CH,
        Hcur /*residual*/, Hcur, f_concat + (size_t)(blk + 1) * CH, cntInvEdge);
  }

  // --- fusion: y = F(16384x896) @ Wf^T(896x1024) via WMMA ---
  cvt_hilo_kernel<<<(EMB_ * FUSE_) / 256, 256, 0, stream>>>(Wf, Wfhi, Wflo, EMB_ * FUSE_);
  gemm_f32a_u16b_kernel<<<((BN_ / 16) * (EMB_ / 16)) / 8, 256, 0, stream>>>(
      f_concat, Wfhi, Wflo, ybuf, BN_, EMB_, FUSE_, FUSE_, FUSE_, EMB_);
  colstats_kernel<<<EMB_, 256, 0, stream>>>(ybuf, statsF, BN_, EMB_);
  fusion_reduce_kernel<<<(BB * EMB_) / 256, 256, 0, stream>>>(ybuf, statsF, gf, betaf,
                                                              zvec, NN, EMB_);

  // --- MLP head (tiny; scalar) ---
  fc_kernel<<<(BB * 512 + 255) / 256, 256, 0, stream>>>(zvec, W1, b1, t1, BB, 2 * EMB_, 512);
  bn_rows_leaky_kernel<<<2, 256, 0, stream>>>(t1, g1, beta1, BB, 512);
  fc_kernel<<<(BB * 256 + 255) / 256, 256, 0, stream>>>(t1, W2, b2, t2, BB, 512, 256);
  bn_rows_leaky_kernel<<<1, 256, 0, stream>>>(t2, g2, beta2, BB, 256);
  // out (second tuple element) -> d_out[512..1024)
  fc_kernel<<<(BB * 64 + 255) / 256, 256, 0, stream>>>(t2, W3, b3, out_f + 512, BB, 256, 64);
  // proj (first tuple element) -> d_out[0..512)
  fc_kernel<<<(BB * 64 + 255) / 256, 256, 0, stream>>>(out_f + 512, Wp, bp, out_f, BB, 64, 64);
}
